// RNNwithAttention_16252156248500
// MI455X (gfx1250) — matrix operand order repaired and checked by reading, not compiled, not run on hardware
//
#include <hip/hip_runtime.h>
#include <hip/hip_bf16.h>

// ---------------------------------------------------------------------------
// Types for CDNA5 WMMA (wave32, 16x16x32 bf16 -> f32)
// ---------------------------------------------------------------------------
typedef __bf16 bf16_t;
typedef __attribute__((ext_vector_type(16))) __bf16 v16bf;
typedef __attribute__((ext_vector_type(8)))  __bf16 v8bf;
typedef __attribute__((ext_vector_type(8)))  float  v8f;

static __device__ __forceinline__ v16bf cat8(v8bf lo, v8bf hi) {
  return __builtin_shufflevector(lo, hi, 0,1,2,3,4,5,6,7,8,9,10,11,12,13,14,15);
}

// A fragment: 16x32 bf16, row-major source with leading dim ld.
static __device__ __forceinline__ v16bf ldA(const bf16_t* A, int ld, int m, int k0, int lane) {
  const bf16_t* p = A + (size_t)m * ld + k0 + ((lane & 16) ? 8 : 0);
  return cat8(*(const v8bf*)p, *(const v8bf*)(p + 16));
}

// B fragment: K x 16 from W stored row-major (N x K) -> B[k][n] = W[n][k].
static __device__ __forceinline__ v16bf ldB(const bf16_t* W, int ld, int n, int k0, int lane) {
  const bf16_t* p = W + (size_t)n * ld + k0 + ((lane & 16) ? 8 : 0);
  return cat8(*(const v8bf*)p, *(const v8bf*)(p + 16));
}

static __device__ __forceinline__ v8f bfwmma(v16bf a, v16bf b, v8f c) {
  return __builtin_amdgcn_wmma_f32_16x16x32_bf16(false, a, false, b, (short)0, c, false, false);
}

static __device__ __forceinline__ float sigf(float x) {
  return 1.0f / (1.0f + __expf(-x));
}
// fast tanh via exp (clamped so exp never overflows)
static __device__ __forceinline__ float ftanh(float x) {
  x = fminf(fmaxf(x, -15.0f), 15.0f);
  float t = __expf(2.0f * x);
  return (t - 1.0f) / (t + 1.0f);
}

// ---------------------------------------------------------------------------
// CDNA5 async global->LDS copy (ASYNCcnt-tracked), one 16B packet per lane.
// ---------------------------------------------------------------------------
static __device__ __forceinline__ void async_b128(void* lds_dst, const void* gsrc) {
  unsigned int ldsa = (unsigned int)(uintptr_t)lds_dst;
  unsigned long long ga = (unsigned long long)(uintptr_t)gsrc;
  asm volatile("global_load_async_to_lds_b128 %0, %1, off"
               :: "v"(ldsa), "v"(ga) : "memory");
}
static __device__ __forceinline__ void async_wait0() {
  asm volatile("s_wait_asynccnt 0" ::: "memory");
}

// ---------------------------------------------------------------------------
// Problem constants
// ---------------------------------------------------------------------------
#define TT 512
#define BB 64
#define DD 512
#define HH 256
#define GG 2048
#define TBm (TT * BB)   // 32768
#define NWG 32          // persistent workgroups in the LSTM scan
#define NTH 512         // threads per workgroup
#define NT  (NWG * NTH)          // 16384 threads grid-wide
#define NWAVES (NWG * (NTH/32))  // 512 waves grid-wide

// ---------------------------------------------------------------------------
// f32 -> bf16 conversion (supports strided 2D source for att_w1 slices)
// ---------------------------------------------------------------------------
__global__ void k_cvt_bf16(const float* __restrict__ src, bf16_t* __restrict__ dst,
                           int rows, int cols, int sld) {
  int i = blockIdx.x * blockDim.x + threadIdx.x;
  if (i >= rows * cols) return;
  int r = i / cols, c = i - r * cols;
  dst[i] = (bf16_t)src[(size_t)r * sld + c];
}

__global__ void k_zero_bar(unsigned* p) {
  if (threadIdx.x < 2) p[threadIdx.x] = 0u;
}

// ---------------------------------------------------------------------------
// Grid-wide sense-reversal barrier (single counter+generation pair in ws).
// ---------------------------------------------------------------------------
static __device__ __forceinline__ void gridbar(unsigned* cnt, unsigned* gen, unsigned nblk) {
  __syncthreads();
  if (threadIdx.x == 0) {
    __threadfence();
    unsigned g = atomicAdd(gen, 0u);
    if (atomicAdd(cnt, 1u) == nblk - 1) {
      atomicExch(cnt, 0u);
      __threadfence();
      atomicAdd(gen, 1u);           // release
    } else {
      while (atomicAdd(gen, 0u) == g) __builtin_amdgcn_s_sleep(2);
      __threadfence();
    }
  }
  __syncthreads();
}

// ---------------------------------------------------------------------------
// Big WMMA GEMM:  C[M,N] = A[M,K] @ W[N,K]^T + bias1 + bias2
// 8 waves/block; wave tile 16x64. K loop is a 2-stage ping-pong pipeline:
// two independent fragment sets alternate, so no rotation copies are needed
// and next-step loads overlap current WMMAs.  Requires K % 64 == 0.
// ---------------------------------------------------------------------------
__global__ __launch_bounds__(256)
void k_gemm_bt(const bf16_t* __restrict__ A, const bf16_t* __restrict__ W,
               const float* __restrict__ bias1, const float* __restrict__ bias2,
               bf16_t* __restrict__ Cb, float* __restrict__ Cf,
               int M, int N, int K) {
  int lane = threadIdx.x & 31;
  int wave = threadIdx.x >> 5;
  int m0 = blockIdx.y * 128 + wave * 16;
  int n0 = blockIdx.x * 64;
  int mrow = m0 + (lane & 15);

  v8f acc[4] = {{}, {}, {}, {}};
  v16bf aA = ldA(A, K, mrow, 0, lane);
  v16bf bA[4];
#pragma unroll
  for (int j = 0; j < 4; ++j) bA[j] = ldB(W, K, n0 + j * 16 + (lane & 15), 0, lane);
  v16bf aB = ldA(A, K, mrow, 32, lane);
  v16bf bB[4];
#pragma unroll
  for (int j = 0; j < 4; ++j) bB[j] = ldB(W, K, n0 + j * 16 + (lane & 15), 32, lane);

  for (int k = 64; k < K; k += 64) {
#pragma unroll
    for (int j = 0; j < 4; ++j) acc[j] = bfwmma(aA, bA[j], acc[j]);
    aA = ldA(A, K, mrow, k, lane);
#pragma unroll
    for (int j = 0; j < 4; ++j) bA[j] = ldB(W, K, n0 + j * 16 + (lane & 15), k, lane);
#pragma unroll
    for (int j = 0; j < 4; ++j) acc[j] = bfwmma(aB, bB[j], acc[j]);
    aB = ldA(A, K, mrow, k + 32, lane);
#pragma unroll
    for (int j = 0; j < 4; ++j) bB[j] = ldB(W, K, n0 + j * 16 + (lane & 15), k + 32, lane);
  }
#pragma unroll
  for (int j = 0; j < 4; ++j) acc[j] = bfwmma(aA, bA[j], acc[j]);
#pragma unroll
  for (int j = 0; j < 4; ++j) acc[j] = bfwmma(aB, bB[j], acc[j]);

  int mb = m0 + ((lane & 16) ? 8 : 0);
#pragma unroll
  for (int j = 0; j < 4; ++j) {
    int ncol = n0 + j * 16 + (lane & 15);
    float bs = (bias1 ? bias1[ncol] : 0.0f) + (bias2 ? bias2[ncol] : 0.0f);
#pragma unroll
    for (int v = 0; v < 8; ++v) {
      float val = acc[j][v] + bs;
      size_t idx = (size_t)(mb + v) * N + ncol;
      if (Cb) Cb[idx] = (bf16_t)val;
      if (Cf) Cf[idx] = val;
    }
  }
}

// ---------------------------------------------------------------------------
// Bi-directional RNN scan. One block per direction. Recurrent h in LDS;
// xp[t] tile double-buffered in LDS via ASYNCcnt-tracked async copies.
// ---------------------------------------------------------------------------
__global__ __launch_bounds__(256)
void k_rnn_scan(const bf16_t* __restrict__ xp_f, const bf16_t* __restrict__ xp_b,
                const bf16_t* __restrict__ whf, const bf16_t* __restrict__ whb,
                bf16_t* __restrict__ h_ctx) {
  __shared__ __attribute__((aligned(16))) bf16_t h[BB * HH];          // 32 KB
  __shared__ __attribute__((aligned(16))) bf16_t xpbuf[2][BB * HH];   // 2 x 32 KB
  int dir = blockIdx.x;
  const bf16_t* xp = dir ? xp_b : xp_f;
  const bf16_t* wh = dir ? whb : whf;
  int tid = threadIdx.x, lane = tid & 31, wave = tid >> 5;

  for (int i = tid; i < BB * HH; i += 256) h[i] = (bf16_t)0.0f;
  {
    int t0 = dir ? (TT - 1) : 0;
    const char* src = (const char*)(xp + (size_t)t0 * BB * HH);
    char* dst = (char*)xpbuf[0];
    for (int i = tid; i < (BB * HH * 2) / 16; i += 256)
      async_b128(dst + i * 16, src + i * 16);
  }
  async_wait0();
  __syncthreads();

  for (int s = 0; s < TT; ++s) {
    int t = dir ? (TT - 1 - s) : s;
    int cur = s & 1;
    if (s + 1 < TT) {
      int tn = dir ? (TT - 2 - s) : (s + 1);
      const char* src = (const char*)(xp + (size_t)tn * BB * HH);
      char* dst = (char*)xpbuf[cur ^ 1];
      for (int i = tid; i < (BB * HH * 2) / 16; i += 256)
        async_b128(dst + i * 16, src + i * 16);
    }

    v8f acc[8];
#pragma unroll
    for (int i = 0; i < 8; ++i) {
      int tile = wave * 8 + i;
      int tm = (tile & 3) * 16, tn = (tile >> 2) * 16;
      v8f a = {};
#pragma unroll 2
      for (int k = 0; k < HH; k += 32)
        a = bfwmma(ldA(h, HH, tm + (lane & 15), k, lane),
                   ldB(wh, HH, tn + (lane & 15), k, lane), a);
      acc[i] = a;
    }
    async_wait0();
    __syncthreads();

    const bf16_t* xpl = xpbuf[cur];
#pragma unroll
    for (int i = 0; i < 8; ++i) {
      int tile = wave * 8 + i;
      int tm = (tile & 3) * 16, tn = (tile >> 2) * 16;
      int n = tn + (lane & 15);
      int mb = tm + ((lane & 16) ? 8 : 0);
#pragma unroll
      for (int v = 0; v < 8; ++v) {
        int m = mb + v;
        float val = acc[i][v] + (float)xpl[m * HH + n];
        bf16_t hv = (bf16_t)ftanh(val);
        h[m * HH + n] = hv;
        h_ctx[((size_t)t * BB + m) * DD + dir * HH + n] = hv;
      }
    }
    __syncthreads();
  }
}

// ---------------------------------------------------------------------------
// Grid-distributed 64xN GEMM used inside the persistent LSTM scan:
//   G[64,N] = A1[64,K1] @ W1[N,K1]^T (+ A2 @ W2^T) + b1 + b2
// One 16x16 tile per (global) wave, strided by total wave count.
// ---------------------------------------------------------------------------
static __device__ __forceinline__
void gemmMW(const bf16_t* A1, const bf16_t* W1, int K1,
            const bf16_t* A2, const bf16_t* W2, int K2,
            const float* b1, const float* b2,
            float* __restrict__ Gout, int N, int gwave, int lane) {
  int ntiles = 4 * (N >> 4);
  for (int tile = gwave; tile < ntiles; tile += NWAVES) {
    int tm = (tile & 3) * 16, tn = (tile >> 2) * 16;
    int mrow = tm + (lane & 15);
    int ncol = tn + (lane & 15);
    v8f acc = {};
#pragma unroll 4
    for (int k = 0; k < K1; k += 32)
      acc = bfwmma(ldA(A1, K1, mrow, k, lane), ldB(W1, K1, ncol, k, lane), acc);
    if (A2) {
#pragma unroll 4
      for (int k = 0; k < K2; k += 32)
        acc = bfwmma(ldA(A2, K2, mrow, k, lane), ldB(W2, K2, ncol, k, lane), acc);
    }
    float bs = (b1 ? b1[ncol] : 0.0f) + (b2 ? b2[ncol] : 0.0f);
    int mb = tm + ((lane & 16) ? 8 : 0);
#pragma unroll
    for (int v = 0; v < 8; ++v)
      Gout[(mb + v) * N + ncol] = acc[v] + bs;
  }
}

// ---------------------------------------------------------------------------
// Persistent attention + 2-layer LSTM scan across NWG workgroups with a
// grid barrier. Every phase of every step is partitioned grid-wide, so the
// 33 MB/step attention reads hit aggregate L2 bandwidth instead of one WGP.
// ---------------------------------------------------------------------------
__global__ __launch_bounds__(NTH)
void k_lstm_scan(const bf16_t* __restrict__ xb,      // (T,B,D) bf16
                 const bf16_t* __restrict__ hctx,    // (T,B,D) bf16
                 const bf16_t* __restrict__ projb,   // (T,B,D) bf16 (incl. att_b1)
                 const bf16_t* __restrict__ wsb,     // (D,D)
                 const float* __restrict__ w2, const float* __restrict__ b2,
                 const bf16_t* __restrict__ wih0, const bf16_t* __restrict__ whh0,
                 const float* __restrict__ bih0, const float* __restrict__ bhh0,
                 const bf16_t* __restrict__ wih1, const bf16_t* __restrict__ whh1,
                 const float* __restrict__ bih1, const float* __restrict__ bhh1,
                 float* __restrict__ q,  float* __restrict__ e,
                 float* __restrict__ g0, float* __restrict__ g1,
                 float* __restrict__ c0, float* __restrict__ c1,
                 float* __restrict__ h0f,
                 bf16_t* __restrict__ h0b, bf16_t* __restrict__ h1b,
                 bf16_t* __restrict__ sb,
                 float* __restrict__ out,
                 unsigned* __restrict__ bar) {
  __shared__ float w2s[DD];     // 2 KB
  __shared__ float red[NTH];    // 2 KB  (per-WG softmax-sum reduction)
  __shared__ float sinv[BB];    // 1/sum per batch column
  int tid = threadIdx.x, lane = tid & 31;
  int gtid = blockIdx.x * NTH + tid;
  int gwave = gtid >> 5;
  unsigned* cnt = bar;
  unsigned* gen = bar + 1;

  for (int i = tid; i < DD; i += NTH) w2s[i] = w2[i];
  for (int i = gtid; i < BB * DD; i += NT) {
    c0[i] = 0.0f; c1[i] = 0.0f; h0f[i] = 0.0f;
    h0b[i] = (bf16_t)0.0f; h1b[i] = (bf16_t)0.0f;
  }
  float b2v = b2[0];
  gridbar(cnt, gen, NWG);

  for (int t = 0; t < TT; ++t) {
    // ---- s = h0 + attention(h0)   (t == 0: s = h0 = 0) ----
    if (t > 0) {
      // q = h0 @ ws^T  (64 x 512), grid-distributed
      gemmMW(h0b, wsb, DD, nullptr, nullptr, 0, nullptr, nullptr, q, DD, gwave, lane);
      gridbar(cnt, gen, NWG);

      // e[t',b] = exp( tanh( sum_d tanh(q[b,d]+proj[t',b,d]) * w2[d] + b2 ) )
      // (logits are tanh-bounded, so exp without max-shift is safe)
      for (int idx = gtid; idx < TT * BB; idx += NT) {
        int tp = idx >> 6, b = idx & 63;
        const bf16_t* ph = projb + ((size_t)tp * BB + b) * DD;
        const float* qb = q + b * DD;
        float acc = 0.0f;
        for (int d = 0; d < DD; d += 8) {
          v8bf pv = *(const v8bf*)(ph + d);
#pragma unroll
          for (int u = 0; u < 8; ++u)
            acc += ftanh(qb[d + u] + (float)pv[u]) * w2s[d + u];
        }
        e[idx] = __expf(ftanh(acc + b2v));
      }
      gridbar(cnt, gen, NWG);

      // per-WG redundant softmax denominators (read-only on e)
      {
        int b = tid & 63, chunk = tid >> 6;   // 8 chunks x 64 timesteps
        float s_ = 0.0f;
        for (int k2 = 0; k2 < 64; ++k2) s_ += e[(chunk * 64 + k2) * BB + b];
        red[tid] = s_;
        __syncthreads();
        if (tid < BB) {
          float ss = 0.0f;
          for (int c = 0; c < 8; ++c) ss += red[c * 64 + tid];
          sinv[tid] = 1.0f / ss;
        }
        __syncthreads();
      }
      // ctx[b,d] = (sum_t e[t,b]*hctx[t,b,d]) * sinv[b];  s = h0 + ctx
      for (int idx = gtid; idx < BB * DD; idx += NT) {
        int b = idx >> 9, d = idx & 511;
        float acc = 0.0f;
        for (int tp = 0; tp < TT; ++tp)
          acc += e[tp * BB + b] * (float)hctx[((size_t)tp * BB + b) * DD + d];
        sb[idx] = (bf16_t)(h0f[idx] + acc * sinv[b]);
      }
      gridbar(cnt, gen, NWG);
    } else {
      for (int idx = gtid; idx < BB * DD; idx += NT) sb[idx] = h0b[idx];
      gridbar(cnt, gen, NWG);
    }

    // ---- layer 0 gates: g0 = x[t] @ wih0^T + s @ whh0^T + bih0 + bhh0 ----
    gemmMW(xb + (size_t)t * BB * DD, wih0, DD, sb, whh0, DD, bih0, bhh0,
           g0, GG, gwave, lane);
    gridbar(cnt, gen, NWG);
    for (int idx = gtid; idx < BB * DD; idx += NT) {
      int b = idx >> 9, j = idx & 511;
      const float* gb = g0 + b * GG;
      float ig = sigf(gb[j]), fg = sigf(gb[512 + j]);
      float gg = ftanh(gb[1024 + j]), og = sigf(gb[1536 + j]);
      float c = fg * c0[idx] + ig * gg;
      c0[idx] = c;
      float hh = og * ftanh(c);
      h0f[idx] = hh; h0b[idx] = (bf16_t)hh;
    }
    gridbar(cnt, gen, NWG);

    // ---- layer 1 gates: g1 = h0 @ wih1^T + h1 @ whh1^T + bih1 + bhh1 ----
    gemmMW(h0b, wih1, DD, h1b, whh1, DD, bih1, bhh1, g1, GG, gwave, lane);
    gridbar(cnt, gen, NWG);
    for (int idx = gtid; idx < BB * DD; idx += NT) {
      int b = idx >> 9, j = idx & 511;
      const float* gb = g1 + b * GG;
      float ig = sigf(gb[j]), fg = sigf(gb[512 + j]);
      float gg = ftanh(gb[1024 + j]), og = sigf(gb[1536 + j]);
      float c = fg * c1[idx] + ig * gg;
      c1[idx] = c;
      float hh = og * ftanh(c);
      h1b[idx] = (bf16_t)hh;
      out[(size_t)t * BB * DD + idx] = hh;
    }
    gridbar(cnt, gen, NWG);
  }
}

// ---------------------------------------------------------------------------
// Host launcher
// ---------------------------------------------------------------------------
extern "C" void kernel_launch(void* const* d_in, const int* in_sizes, int n_in,
                              void* d_out, int out_size, void* d_ws, size_t ws_size,
                              hipStream_t stream) {
  (void)in_sizes; (void)n_in; (void)out_size; (void)ws_size;

  const float* x      = (const float*)d_in[0];
  const float* wif    = (const float*)d_in[1];
  const float* whf    = (const float*)d_in[2];
  const float* bif    = (const float*)d_in[3];
  const float* bhf    = (const float*)d_in[4];
  const float* wib    = (const float*)d_in[5];
  const float* whb    = (const float*)d_in[6];
  const float* bib    = (const float*)d_in[7];
  const float* bhb    = (const float*)d_in[8];
  const float* att_w1 = (const float*)d_in[9];
  const float* att_b1 = (const float*)d_in[10];
  const float* att_w2 = (const float*)d_in[11];
  const float* att_b2 = (const float*)d_in[12];
  const float* l0_wih = (const float*)d_in[13];
  const float* l0_whh = (const float*)d_in[14];
  const float* l0_bih = (const float*)d_in[15];
  const float* l0_bhh = (const float*)d_in[16];
  const float* l1_wih = (const float*)d_in[17];
  const float* l1_whh = (const float*)d_in[18];
  const float* l1_bih = (const float*)d_in[19];
  const float* l1_bhh = (const float*)d_in[20];
  float* out = (float*)d_out;

  char* wsc = (char*)d_ws;
  size_t off = 0;
  auto alloc = [&](size_t bytes) -> void* {
    void* p = wsc + off;
    off = (off + bytes + 255) & ~(size_t)255;
    return p;
  };

  bf16_t* xb      = (bf16_t*)alloc((size_t)TBm * DD * 2);
  bf16_t* wif_b   = (bf16_t*)alloc((size_t)HH * DD * 2);
  bf16_t* whf_b   = (bf16_t*)alloc((size_t)HH * HH * 2);
  bf16_t* wib_b   = (bf16_t*)alloc((size_t)HH * DD * 2);
  bf16_t* whb_b   = (bf16_t*)alloc((size_t)HH * HH * 2);
  bf16_t* ws_b    = (bf16_t*)alloc((size_t)DD * DD * 2);
  bf16_t* wh_b    = (bf16_t*)alloc((size_t)DD * DD * 2);
  bf16_t* wih0_b  = (bf16_t*)alloc((size_t)GG * DD * 2);
  bf16_t* whh0_b  = (bf16_t*)alloc((size_t)GG * DD * 2);
  bf16_t* wih1_b  = (bf16_t*)alloc((size_t)GG * DD * 2);
  bf16_t* whh1_b  = (bf16_t*)alloc((size_t)GG * DD * 2);
  bf16_t* xp_f    = (bf16_t*)alloc((size_t)TBm * HH * 2);
  bf16_t* xp_b    = (bf16_t*)alloc((size_t)TBm * HH * 2);
  bf16_t* hctx_b  = (bf16_t*)alloc((size_t)TBm * DD * 2);
  bf16_t* proj_b  = (bf16_t*)alloc((size_t)TBm * DD * 2);
  float*  q   = (float*)alloc((size_t)BB * DD * 4);
  float*  e   = (float*)alloc((size_t)TT * BB * 4);
  float*  g0  = (float*)alloc((size_t)BB * GG * 4);
  float*  g1  = (float*)alloc((size_t)BB * GG * 4);
  float*  c0  = (float*)alloc((size_t)BB * DD * 4);
  float*  c1  = (float*)alloc((size_t)BB * DD * 4);
  float*  h0f = (float*)alloc((size_t)BB * DD * 4);
  bf16_t* h0b = (bf16_t*)alloc((size_t)BB * DD * 2);
  bf16_t* h1b = (bf16_t*)alloc((size_t)BB * DD * 2);
  bf16_t* sb  = (bf16_t*)alloc((size_t)BB * DD * 2);
  unsigned* bar = (unsigned*)alloc(2 * sizeof(unsigned));

  auto cvt = [&](const float* src, bf16_t* dst, int rows, int cols, int sld) {
    int n = rows * cols;
    k_cvt_bf16<<<dim3((n + 255) / 256), dim3(256), 0, stream>>>(src, dst, rows, cols, sld);
  };

  // --- precision conversion: f32 -> bf16 ---
  cvt(x,      xb,     TBm, DD, DD);
  cvt(wif,    wif_b,  HH,  DD, DD);
  cvt(whf,    whf_b,  HH,  HH, HH);
  cvt(wib,    wib_b,  HH,  DD, DD);
  cvt(whb,    whb_b,  HH,  HH, HH);
  cvt(att_w1,        ws_b, DD, DD, 2 * DD);   // ws = att_w1[:, :D]
  cvt(att_w1 + DD,   wh_b, DD, DD, 2 * DD);   // wh = att_w1[:, D:]
  cvt(l0_wih, wih0_b, GG, DD, DD);
  cvt(l0_whh, whh0_b, GG, DD, DD);
  cvt(l1_wih, wih1_b, GG, DD, DD);
  cvt(l1_whh, whh1_b, GG, DD, DD);

  // --- RNN input projections: xp = x @ wi^T + bi + bh ---
  k_gemm_bt<<<dim3(HH / 64, TBm / 128), dim3(256), 0, stream>>>(
      xb, wif_b, bif, bhf, xp_f, nullptr, TBm, HH, DD);
  k_gemm_bt<<<dim3(HH / 64, TBm / 128), dim3(256), 0, stream>>>(
      xb, wib_b, bib, bhb, xp_b, nullptr, TBm, HH, DD);

  // --- bidirectional RNN scan -> h_ctx ---
  k_rnn_scan<<<dim3(2), dim3(256), 0, stream>>>(xp_f, xp_b, whf_b, whb_b, hctx_b);

  // --- proj_h = h_ctx @ wh^T + att_b1 ---
  k_gemm_bt<<<dim3(DD / 64, TBm / 128), dim3(256), 0, stream>>>(
      hctx_b, wh_b, att_b1, nullptr, proj_b, nullptr, TBm, DD, DD);

  // --- persistent multi-WG attention + LSTM scan ---
  k_zero_bar<<<dim3(1), dim3(32), 0, stream>>>(bar);
  k_lstm_scan<<<dim3(NWG), dim3(NTH), 0, stream>>>(
      xb, hctx_b, proj_b, ws_b, att_w2, att_b2,
      wih0_b, whh0_b, l0_bih, l0_bhh,
      wih1_b, whh1_b, l1_bih, l1_bhh,
      q, e, g0, g1, c0, c1, h0f, h0b, h1b, sb, out, bar);
}
